// Block_91190745628737
// MI455X (gfx1250) — matrix operand order repaired and checked by reading, not compiled, not run on hardware
//
#include <hip/hip_runtime.h>
#include <hip/hip_bf16.h>
#include <math.h>

// ---------------------------------------------------------------------------
// Types matching the CDNA5 WMMA builtin signatures
// ---------------------------------------------------------------------------
typedef __attribute__((ext_vector_type(16))) __bf16 v16bf;
typedef __attribute__((ext_vector_type(8)))  __bf16 v8bf;
typedef __attribute__((ext_vector_type(8)))  float  v8f;

#define DIMC 256
#define DIC  512          // 2*DIM
#define LQ   17
#define BQ   4096
#define RROWS (BQ * LQ)   // 69632 token rows

// ---------------------------------------------------------------------------
// Graph / permutation constants
// ---------------------------------------------------------------------------
__constant__ int c_HOP[17]   = {0,1,4,7,2,5,8,3,6,9,11,14,10,12,15,13,16};
__constant__ int c_BPIDX[17] = {0,1,2,0,1,2,0,1,2,0,3,4,0,3,4,3,4};
// inverse of GRAPH (so writing seq-row t lands on graph-row GINV[t])
__constant__ int c_GINV[17]  = {0,1,4,7,2,5,8,3,6,9,11,14,10,12,15,13,16};
// adjacency (incl. self loops) as bitmasks + degrees (for A_HAT = D^-1/2 A D^-1/2)
__constant__ unsigned c_ADJ[17] = {0x93u,0x7u,0xEu,0xCu,0x31u,0x70u,0x60u,0x181u,
                                   0x4B80u,0x700u,0x600u,0x1900u,0x3800u,0x3000u,
                                   0xC100u,0x1C000u,0x18000u};
__constant__ float c_DEG[17] = {4,3,3,2,3,3,2,3,5,3,2,3,3,2,3,3,2};

__device__ inline float gelu_tanh(float x) {
  float x3 = x * x * x;
  return 0.5f * x * (1.f + tanhf(0.7978845608028654f * (x + 0.044715f * x3)));
}
__device__ inline float siluf(float x) { return x / (1.f + expf(-x)); }
__device__ inline float softplusf(float x) { return (x > 20.f) ? x : log1pf(expf(x)); }

__device__ inline v8f wmma_bf16(v16bf a, v16bf b, v8f c) {
  return __builtin_amdgcn_wmma_f32_16x16x32_bf16(false, a, false, b, (short)0, c,
                                                 false, false);
}

// ---------------------------------------------------------------------------
// Weight transpose + f32 -> bf16 convert: Wt[n*K+k] = W[k*N+n]
// ---------------------------------------------------------------------------
__global__ __launch_bounds__(256) void transpose_w(const float* __restrict__ W,
                                                   __bf16* __restrict__ Wt,
                                                   int K, int N) {
  int t = blockIdx.x * 256 + threadIdx.x;
  if (t < K * N) {
    int k = t / N, n = t % N;
    Wt[(size_t)n * K + k] = (__bf16)W[t];
  }
}

// ---------------------------------------------------------------------------
// bf16 WMMA GEMM:  C[M,N] = A[M,K] @ Bt[N,K]^T, epilogue fused.
// One wave -> 16x64 output strip (4 accumulators). blockDim = (32,8).
// EPI: 0 = bf16 out
//      1 = bf16( gelu(acc + bias) )
//      2 = f32  acc + bias + res1
//      3 = f32  acc + bias + res1 + res2
//      4 = f32  permuted: out[b,GINV[t]] = acc + 2*res1[b,GINV[t]]
// ---------------------------------------------------------------------------
template <int EPI>
__global__ __launch_bounds__(256) void gemm_bf16_wmma(
    const __bf16* __restrict__ A, const __bf16* __restrict__ Bt,
    int M, int N, int K,
    const float* __restrict__ bias, const float* __restrict__ res1,
    const float* __restrict__ res2, float* __restrict__ outF,
    __bf16* __restrict__ outB) {
  const int lane = threadIdx.x;
  const int wave = threadIdx.y;
  const int ng   = N >> 6;                      // groups of 64 columns
  const int tile = blockIdx.x * 8 + wave;
  const int mt = tile / ng, n4 = tile % ng;
  const int mbase = mt * 16, nbase = n4 * 64;
  const int lo = lane & 15, half = lane >> 4;

  const __bf16* pa = A + (size_t)(mbase + lo) * K;
  const __bf16* pb = Bt + (size_t)(nbase + lo) * K;

  v8f acc[4] = {v8f{}, v8f{}, v8f{}, v8f{}};

  for (int k0 = 0; k0 < K; k0 += 32) {
    // A fragment: lanes 0-15 hold K = 0..15; lanes 16-31 K = 16..31
    v8bf alo = *(const v8bf*)(pa + k0 + half * 16);
    v8bf ahi = *(const v8bf*)(pa + k0 + 8 + half * 16);
    v16bf afrag = __builtin_shufflevector(alo, ahi, 0, 1, 2, 3, 4, 5, 6, 7, 8, 9,
                                          10, 11, 12, 13, 14, 15);
    // B fragments from transposed weight: lane = column, K = 16*half + 0..15
    const __bf16* pbk = pb + k0 + 16 * half;
#pragma unroll
    for (int j = 0; j < 4; ++j) {
      v16bf bfrag = *(const v16bf*)(pbk + (size_t)j * 16 * K);
      acc[j] = wmma_bf16(afrag, bfrag, acc[j]);
    }
  }

#pragma unroll
  for (int j = 0; j < 4; ++j) {
    const int col = nbase + j * 16 + lo;
    float bv = 0.f;
    if (EPI == 1 || EPI == 2 || EPI == 3) bv = bias[col];
#pragma unroll
    for (int v = 0; v < 8; ++v) {
      const int row = mbase + half * 8 + v;
      float val = acc[j][v];
      if (EPI == 0) {
        outB[(size_t)row * N + col] = (__bf16)val;
      } else if (EPI == 1) {
        outB[(size_t)row * N + col] = (__bf16)gelu_tanh(val + bv);
      } else if (EPI == 2) {
        size_t o = (size_t)row * N + col;
        outF[o] = val + bv + res1[o];
      } else if (EPI == 3) {
        size_t o = (size_t)row * N + col;
        outF[o] = val + bv + res1[o] + res2[o];
      } else if (EPI == 4) {
        int bb = row / 17, t = row % 17;
        size_t o = (size_t)(bb * 17 + c_GINV[t]) * N + col;
        outF[o] = val + 2.f * res1[o];
      }
    }
  }
}

// ---------------------------------------------------------------------------
// LayerNorm over 256 dims, one wave per row, fp32 in -> bf16 out
// ---------------------------------------------------------------------------
__global__ __launch_bounds__(256) void ln_rows(const float* __restrict__ x,
                                               const float* __restrict__ g,
                                               const float* __restrict__ bb,
                                               __bf16* __restrict__ out) {
  const int row = blockIdx.x * 8 + threadIdx.y;
  const int lane = threadIdx.x;
  const float* p = x + (size_t)row * DIMC + lane * 8;
  float4 q0 = *(const float4*)p;
  float4 q1 = *(const float4*)(p + 4);
  float v[8] = {q0.x, q0.y, q0.z, q0.w, q1.x, q1.y, q1.z, q1.w};
  float s = 0.f;
#pragma unroll
  for (int i = 0; i < 8; ++i) s += v[i];
  for (int o = 16; o > 0; o >>= 1) s += __shfl_xor(s, o, 32);
  float mean = s * (1.f / DIMC);
  float s2 = 0.f;
#pragma unroll
  for (int i = 0; i < 8; ++i) { float d = v[i] - mean; s2 += d * d; }
  for (int o = 16; o > 0; o >>= 1) s2 += __shfl_xor(s2, o, 32);
  float inv = rsqrtf(s2 * (1.f / DIMC) + 1e-5f);
#pragma unroll
  for (int i = 0; i < 8; ++i) {
    int d = lane * 8 + i;
    out[(size_t)row * DIMC + d] = (__bf16)((v[i] - mean) * inv * g[d] + bb[d]);
  }
}

// LN variant for the mamba input: gather x[b, HOP[t]] + bpe[BPIDX[t]], then LN
__global__ __launch_bounds__(256) void ln_hop_rows(const float* __restrict__ x,
                                                   const float* __restrict__ bpe,
                                                   const float* __restrict__ g,
                                                   const float* __restrict__ bb,
                                                   __bf16* __restrict__ out) {
  const int row = blockIdx.x * 8 + threadIdx.y;
  const int lane = threadIdx.x;
  const int b = row / 17, t = row % 17;
  const float* p = x + ((size_t)b * 17 + c_HOP[t]) * DIMC + lane * 8;
  const float* pb = bpe + (size_t)c_BPIDX[t] * DIMC + lane * 8;
  float4 q0 = *(const float4*)p;
  float4 q1 = *(const float4*)(p + 4);
  float4 e0 = *(const float4*)pb;
  float4 e1 = *(const float4*)(pb + 4);
  float v[8] = {q0.x + e0.x, q0.y + e0.y, q0.z + e0.z, q0.w + e0.w,
                q1.x + e1.x, q1.y + e1.y, q1.z + e1.z, q1.w + e1.w};
  float s = 0.f;
#pragma unroll
  for (int i = 0; i < 8; ++i) s += v[i];
  for (int o = 16; o > 0; o >>= 1) s += __shfl_xor(s, o, 32);
  float mean = s * (1.f / DIMC);
  float s2 = 0.f;
#pragma unroll
  for (int i = 0; i < 8; ++i) { float d = v[i] - mean; s2 += d * d; }
  for (int o = 16; o > 0; o >>= 1) s2 += __shfl_xor(s2, o, 32);
  float inv = rsqrtf(s2 * (1.f / DIMC) + 1e-5f);
#pragma unroll
  for (int i = 0; i < 8; ++i) {
    int d = lane * 8 + i;
    out[(size_t)row * DIMC + d] = (__bf16)((v[i] - mean) * inv * g[d] + bb[d]);
  }
}

// ---------------------------------------------------------------------------
// Graph mixing: out[b,i,d] = sum_j A_HAT[i,j] * in[b,j,d]   (bf16 -> bf16)
// ---------------------------------------------------------------------------
__global__ __launch_bounds__(256) void graphmix(const __bf16* __restrict__ in,
                                                __bf16* __restrict__ out, int W) {
  const int tid = blockIdx.x * 256 + threadIdx.x;
  const int b = tid / W, d = tid % W;
  float vals[17];
#pragma unroll
  for (int j = 0; j < 17; ++j)
    vals[j] = (float)in[((size_t)b * 17 + j) * W + d];
#pragma unroll
  for (int i = 0; i < 17; ++i) {
    float acc = 0.f;
    unsigned m = c_ADJ[i];
    float di = c_DEG[i];
#pragma unroll
    for (int j = 0; j < 17; ++j)
      if ((m >> j) & 1u) acc += rsqrtf(di * c_DEG[j]) * vals[j];
    out[((size_t)b * 17 + i) * W + d] = (__bf16)acc;
  }
}

// ---------------------------------------------------------------------------
// Mamba conv + silu:  xm = silu(xz[:, :512] * conv_w + conv_b)
// ---------------------------------------------------------------------------
__global__ __launch_bounds__(256) void conv_silu(const __bf16* __restrict__ xz,
                                                 const float* __restrict__ cw,
                                                 const float* __restrict__ cb,
                                                 __bf16* __restrict__ xm) {
  const int tid = blockIdx.x * 256 + threadIdx.x;
  const int r = tid >> 9, d = tid & 511;
  float v = (float)xz[(size_t)r * 1024 + d] * cw[d] + cb[d];
  xm[tid] = (__bf16)siluf(v);
}

// ---------------------------------------------------------------------------
// x_proj: dbl[r, 0:24] = xm[r, 0:512] @ x_proj_w(512x24). One wave per row.
// ---------------------------------------------------------------------------
__global__ __launch_bounds__(256) void xproj24(const __bf16* __restrict__ xm,
                                               const float* __restrict__ w,
                                               float* __restrict__ dbl) {
  __shared__ float sw[512 * 24];
  const int tf = threadIdx.y * 32 + threadIdx.x;
  for (int i = tf; i < 512 * 24; i += 256) sw[i] = w[i];
  __syncthreads();
  const int row = blockIdx.x * 8 + threadIdx.y;
  const int lane = threadIdx.x;
  float xv[16];
  const __bf16* p = xm + (size_t)row * 512 + lane * 16;
#pragma unroll
  for (int i = 0; i < 16; ++i) xv[i] = (float)p[i];
  for (int n = 0; n < 24; ++n) {
    float acc = 0.f;
#pragma unroll
    for (int i = 0; i < 16; ++i) acc += xv[i] * sw[(lane * 16 + i) * 24 + n];
    for (int o = 16; o > 0; o >>= 1) acc += __shfl_xor(acc, o, 32);
    if (lane == 0) dbl[(size_t)row * 24 + n] = acc;
  }
}

// ---------------------------------------------------------------------------
// Mamba selective scan. One block per batch element; 2 channels per thread.
// Fuses dt_proj + softplus, the 17-step recurrence, Dp skip and silu(z) gate.
// ---------------------------------------------------------------------------
__global__ __launch_bounds__(256) void mamba_scan(
    const __bf16* __restrict__ xm, const __bf16* __restrict__ xz,
    const float* __restrict__ dbl, const float* __restrict__ dtw,
    const float* __restrict__ dtb, const float* __restrict__ alog,
    const float* __restrict__ dp, __bf16* __restrict__ y) {
  const int b = blockIdx.x;
  __shared__ float sdbl[17][24];
  for (int i = threadIdx.x; i < 17 * 24; i += 256)
    sdbl[i / 24][i % 24] = dbl[(size_t)(b * 17 + i / 24) * 24 + (i % 24)];
  __syncthreads();

  const int d0 = threadIdx.x, d1 = threadIdx.x + 256;
  float w0[16], w1[16];
#pragma unroll
  for (int r = 0; r < 16; ++r) { w0[r] = dtw[r * 512 + d0]; w1[r] = dtw[r * 512 + d1]; }
  float A0[4], A1[4], h0[4] = {0, 0, 0, 0}, h1[4] = {0, 0, 0, 0};
#pragma unroll
  for (int s = 0; s < 4; ++s) {
    A0[s] = -expf(alog[d0 * 4 + s]);
    A1[s] = -expf(alog[d1 * 4 + s]);
  }
  const float bias0 = dtb[d0], bias1 = dtb[d1];
  const float D0 = dp[d0], D1 = dp[d1];

  for (int t = 0; t < 17; ++t) {
    const size_t row = (size_t)b * 17 + t;
    const float x0 = (float)xm[row * 512 + d0];
    const float x1 = (float)xm[row * 512 + d1];
    float dt0 = bias0, dt1 = bias1;
#pragma unroll
    for (int r = 0; r < 16; ++r) {
      dt0 += sdbl[t][r] * w0[r];
      dt1 += sdbl[t][r] * w1[r];
    }
    dt0 = softplusf(dt0);
    dt1 = softplusf(dt1);
    float y0 = 0.f, y1 = 0.f;
#pragma unroll
    for (int s = 0; s < 4; ++s) {
      const float Bm = sdbl[t][16 + s], Cm = sdbl[t][20 + s];
      h0[s] = expf(dt0 * A0[s]) * h0[s] + dt0 * Bm * x0;
      h1[s] = expf(dt1 * A1[s]) * h1[s] + dt1 * Bm * x1;
      y0 += h0[s] * Cm;
      y1 += h1[s] * Cm;
    }
    y0 += D0 * x0;
    y1 += D1 * x1;
    const float z0 = (float)xz[row * 1024 + 512 + d0];
    const float z1 = (float)xz[row * 1024 + 512 + d1];
    y[row * 512 + d0] = (__bf16)(y0 * siluf(z0));
    y[row * 512 + d1] = (__bf16)(y1 * siluf(z1));
  }
}

// ---------------------------------------------------------------------------
// Attention over L=17, head dim 32. One wave per (b, head); 8 heads per block.
// ---------------------------------------------------------------------------
__global__ __launch_bounds__(256) void attn17(const __bf16* __restrict__ qkv,
                                              __bf16* __restrict__ o) {
  const int b = blockIdx.x;
  const int h = threadIdx.y;
  const int e = threadIdx.x;
  __shared__ float sq[8][17][32], sk[8][17][32], sv[8][17][32], ss[8][17][18];

  for (int t = 0; t < 17; ++t) {
    size_t base = (size_t)(b * 17 + t) * 768 + h * 32 + e;
    sq[h][t][e] = (float)qkv[base];
    sk[h][t][e] = (float)qkv[base + 256];
    sv[h][t][e] = (float)qkv[base + 512];
  }
  __syncthreads();

  for (int p = e; p < 289; p += 32) {
    int t = p / 17, u = p % 17;
    float d = 0.f;
#pragma unroll
    for (int i = 0; i < 32; ++i) d += sq[h][t][i] * sk[h][u][i];
    ss[h][t][u] = d * 0.17677669529663687f;  // 32^-0.5
  }
  __syncthreads();

  if (e < 17) {
    float mx = -1e30f;
    for (int u = 0; u < 17; ++u) mx = fmaxf(mx, ss[h][e][u]);
    float sum = 0.f;
    for (int u = 0; u < 17; ++u) {
      float p = expf(ss[h][e][u] - mx);
      ss[h][e][u] = p;
      sum += p;
    }
    float inv = 1.f / sum;
    for (int u = 0; u < 17; ++u) ss[h][e][u] *= inv;
  }
  __syncthreads();

  for (int t = 0; t < 17; ++t) {
    float acc = 0.f;
#pragma unroll
    for (int u = 0; u < 17; ++u) acc += ss[h][t][u] * sv[h][u][e];
    o[(size_t)(b * 17 + t) * 256 + h * 32 + e] = (__bf16)acc;
  }
}

// ---------------------------------------------------------------------------
// Orchestration
// ---------------------------------------------------------------------------
extern "C" void kernel_launch(void* const* d_in, const int* in_sizes, int n_in,
                              void* d_out, int out_size, void* d_ws, size_t ws_size,
                              hipStream_t stream) {
  const float* x        = (const float*)d_in[0];
  const float* gcn_ln_g = (const float*)d_in[1];
  const float* gcn_ln_b = (const float*)d_in[2];
  const float* gcn_w1   = (const float*)d_in[3];
  const float* gcn_b1   = (const float*)d_in[4];
  const float* gcn_w2   = (const float*)d_in[5];
  const float* gcn_b2   = (const float*)d_in[6];
  const float* bp_embed = (const float*)d_in[7];
  const float* ssm_ln_g = (const float*)d_in[8];
  const float* ssm_ln_b = (const float*)d_in[9];
  const float* in_proj_w  = (const float*)d_in[10];
  const float* conv_w     = (const float*)d_in[11];
  const float* conv_b     = (const float*)d_in[12];
  const float* x_proj_w   = (const float*)d_in[13];
  const float* dt_proj_w  = (const float*)d_in[14];
  const float* dt_proj_b  = (const float*)d_in[15];
  const float* A_log      = (const float*)d_in[16];
  const float* Dp         = (const float*)d_in[17];
  const float* out_proj_w = (const float*)d_in[18];
  const float* ln1_g = (const float*)d_in[19];
  const float* ln1_b = (const float*)d_in[20];
  const float* qkv_w = (const float*)d_in[21];
  const float* attn_proj_w = (const float*)d_in[22];
  const float* attn_proj_b = (const float*)d_in[23];
  const float* ln2_g = (const float*)d_in[24];
  const float* ln2_b = (const float*)d_in[25];
  const float* mlp_w1 = (const float*)d_in[26];
  const float* mlp_b1 = (const float*)d_in[27];
  const float* mlp_w2 = (const float*)d_in[28];
  const float* mlp_b2 = (const float*)d_in[29];
  float* out = (float*)d_out;

  const int R = RROWS;
  char* base = (char*)d_ws;
  size_t off = 0;
  auto alloc = [&](size_t bytes) -> char* {
    char* p = base + off;
    off += (bytes + 255) & ~(size_t)255;
    return p;
  };

  // transposed bf16 weights (all L2-resident)
  __bf16* wt_gcn1  = (__bf16*)alloc((size_t)512 * 256 * 2);
  __bf16* wt_gcn2  = (__bf16*)alloc((size_t)256 * 512 * 2);
  __bf16* wt_inpj  = (__bf16*)alloc((size_t)1024 * 256 * 2);
  __bf16* wt_outpj = (__bf16*)alloc((size_t)256 * 512 * 2);
  __bf16* wt_qkv   = (__bf16*)alloc((size_t)768 * 256 * 2);
  __bf16* wt_attnp = (__bf16*)alloc((size_t)256 * 256 * 2);
  __bf16* wt_mlp1  = (__bf16*)alloc((size_t)1024 * 256 * 2);
  __bf16* wt_mlp2  = (__bf16*)alloc((size_t)256 * 1024 * 2);

  // activation scratch (reused across phases)
  __bf16* bfA  = (__bf16*)alloc((size_t)R * 1024 * 2);  // xz -> qkv -> mlp hidden
  __bf16* bfB1 = (__bf16*)alloc((size_t)R * 256 * 2);   // LN outputs
  __bf16* bfB2 = (__bf16*)alloc((size_t)R * 256 * 2);   // gmix1 -> attn out
  __bf16* bfB3 = (__bf16*)alloc((size_t)R * 512 * 2);   // gcn hidden -> xm
  __bf16* bfB4 = (__bf16*)alloc((size_t)R * 512 * 2);   // gmix2 -> scan y
  float*  fC   = (float*)alloc((size_t)R * 256 * 4);    // X1 (post-GCN) -> x1
  float*  fD   = (float*)alloc((size_t)R * 256 * 4);    // X2 (post-mamba)
  float*  fDbl = (float*)alloc((size_t)R * 24 * 4);     // dbl

  auto T = [&](const float* W, __bf16* Wt, int K, int N) {
    int n = K * N;
    transpose_w<<<dim3((n + 255) / 256), dim3(256), 0, stream>>>(W, Wt, K, N);
  };
  T(gcn_w1, wt_gcn1, 256, 512);
  T(gcn_w2, wt_gcn2, 512, 256);
  T(in_proj_w, wt_inpj, 256, 1024);
  T(out_proj_w, wt_outpj, 512, 256);
  T(qkv_w, wt_qkv, 256, 768);
  T(attn_proj_w, wt_attnp, 256, 256);
  T(mlp_w1, wt_mlp1, 256, 1024);
  T(mlp_w2, wt_mlp2, 1024, 256);

  const dim3 gblk(32, 8);
  auto ggrid = [&](int M, int N) { return dim3((unsigned)(((M / 16) * (N / 64)) / 8)); };
  const dim3 lnGrid(R / 8), lnBlk(32, 8);

  // ---- GCN block ----
  ln_rows<<<lnGrid, lnBlk, 0, stream>>>(x, gcn_ln_g, gcn_ln_b, bfB1);
  graphmix<<<dim3(BQ * 256 / 256), dim3(256), 0, stream>>>(bfB1, bfB2, 256);
  gemm_bf16_wmma<1><<<ggrid(R, 512), gblk, 0, stream>>>(
      bfB2, wt_gcn1, R, 512, 256, gcn_b1, nullptr, nullptr, nullptr, bfB3);
  graphmix<<<dim3(BQ * 512 / 256), dim3(256), 0, stream>>>(bfB3, bfB4, 512);
  gemm_bf16_wmma<2><<<ggrid(R, 256), gblk, 0, stream>>>(
      bfB4, wt_gcn2, R, 256, 512, gcn_b2, x, nullptr, fC, nullptr);  // X1

  // ---- Mamba block ----
  ln_hop_rows<<<lnGrid, lnBlk, 0, stream>>>(fC, bp_embed, ssm_ln_g, ssm_ln_b, bfB1);
  gemm_bf16_wmma<0><<<ggrid(R, 1024), gblk, 0, stream>>>(
      bfB1, wt_inpj, R, 1024, 256, nullptr, nullptr, nullptr, nullptr, bfA);  // xz
  conv_silu<<<dim3(R * 512 / 256), dim3(256), 0, stream>>>(bfA, conv_w, conv_b, bfB3);
  xproj24<<<dim3(R / 8), dim3(32, 8), 0, stream>>>(bfB3, x_proj_w, fDbl);
  mamba_scan<<<dim3(BQ), dim3(256), 0, stream>>>(bfB3, bfA, fDbl, dt_proj_w,
                                                 dt_proj_b, A_log, Dp, bfB4);
  gemm_bf16_wmma<4><<<ggrid(R, 256), gblk, 0, stream>>>(
      bfB4, wt_outpj, R, 256, 512, nullptr, fC, nullptr, fD, nullptr);  // X2 = 2*X1 + xs

  // ---- Attention block ----
  ln_rows<<<lnGrid, lnBlk, 0, stream>>>(fD, ln1_g, ln1_b, bfB1);
  gemm_bf16_wmma<0><<<ggrid(R, 768), gblk, 0, stream>>>(
      bfB1, wt_qkv, R, 768, 256, nullptr, nullptr, nullptr, nullptr, bfA);  // qkv
  attn17<<<dim3(BQ), dim3(32, 8), 0, stream>>>(bfA, bfB2);
  gemm_bf16_wmma<2><<<ggrid(R, 256), gblk, 0, stream>>>(
      bfB2, wt_attnp, R, 256, 256, attn_proj_b, fD, nullptr, fC, nullptr);  // x1

  // ---- MLP block + final residuals: out = x1 + mlp + X2 ----
  ln_rows<<<lnGrid, lnBlk, 0, stream>>>(fC, ln2_g, ln2_b, bfB1);
  gemm_bf16_wmma<1><<<ggrid(R, 1024), gblk, 0, stream>>>(
      bfB1, wt_mlp1, R, 1024, 256, mlp_b1, nullptr, nullptr, nullptr, bfA);
  gemm_bf16_wmma<3><<<ggrid(R, 256), gblk, 0, stream>>>(
      bfA, wt_mlp2, R, 256, 1024, mlp_b2, fC, fD, out, nullptr);
}
